// TransformerBlock_38388417691714
// MI455X (gfx1250) — compile-verified, matrix operand order checked by reading, not run on hardware
//
// CDNA5 (gfx1250) dual-stream transformer block, round 4.
// - All dense math via v_wmma_f32_16x16x32_bf16 (bf16 operands, fp32 accum).
// - GEMM: 128x128 block tile, 8 waves (4M x 2N), 32x64 wave tile ->
//   8 WMMA per 6 fragment loads per K-step (raises WMMA:LDS-read ratio).
// - GEMM tiles double-buffered in LDS via global_load_async_to_lds_b128
//   (ASYNCcnt path), overlapping copies with the WMMA pipe.
// - Native (__bf16) casts -> v_cvt_pk_bf16_f32 hardware conversion.
// - Dead fp32 outputs eliminated (attention + FFN1 write bf16 mirrors only).
// - Flash attention: no materialized scores, online softmax, packed LDS stores.
#include <hip/hip_runtime.h>
#include <math.h>

typedef __bf16 v16bf __attribute__((ext_vector_type(16)));
typedef float  v8f   __attribute__((ext_vector_type(8)));

__device__ __forceinline__ __bf16 f2bf(float f) { return (__bf16)f; }
__device__ __forceinline__ unsigned short f2bfu(float f) {
  union { __bf16 b; unsigned short s; } o; o.b = (__bf16)f; return o.s;
}
__device__ __forceinline__ unsigned pack2bf(float lo, float hi) {
  union { __bf16 b[2]; unsigned u; } o;
  o.b[0] = (__bf16)lo; o.b[1] = (__bf16)hi;
  return o.u;
}

// Async global->LDS copy (CDNA5, tracked by ASYNCcnt). 16B per lane.
__device__ __forceinline__ void async_b128(unsigned lds_off, const void* g) {
  asm volatile("global_load_async_to_lds_b128 %0, %1, off"
               :: "v"(lds_off), "v"(g) : "memory");
}
__device__ __forceinline__ void async_wait0() {
  asm volatile("s_wait_asynccnt 0x0" ::: "memory");
}

// Load a 16x32 bf16 WMMA operand from a row-major LDS tile (row stride LDP).
// A-operand layout (ISA 7.12.2): lane m=l&15 holds row m; lane-group g=l>>4
// covers K = {8g..8g+7, 16+8g..16+8g+7}. A row-major tile of B^T (column n of
// B stored as row n) feeds the B operand with the identical pattern.
template <int LDP>
__device__ __forceinline__ v16bf load_frag_rm(const __bf16* base, int lane) {
  const int m = lane & 15, g = lane >> 4;
  const __bf16* p = base + m * LDP + g * 8;
  v16bf f;
#pragma unroll
  for (int e = 0; e < 8; ++e) f[e] = p[e];
#pragma unroll
  for (int e = 0; e < 8; ++e) f[e + 8] = p[16 + e];
  return f;
}

__device__ __forceinline__ v8f wmma_bf16(v16bf a, v16bf b, v8f c) {
  return __builtin_amdgcn_wmma_f32_16x16x32_bf16(false, a, false, b, (short)0, c,
                                                 false, false);
}

// ---------------------------------------------------------------------------
// C[M,N] = act(Abf[M,K] @ Wtbf[N,K]^T + bias) + resid
// Optional fp32 output (CF32) and bf16 mirror (CBF).
// Block tile 128x128, 8 waves (4 M x 2 N), wave tile 32x64, K-step 32.
// Double-buffered LDS; tiles staged with async b128 (4 chunks/thread).
// ---------------------------------------------------------------------------
template <bool BIAS, bool RESID, bool GELU, bool CF32, bool CBF>
__global__ __launch_bounds__(256) void gemm_bf16_kernel(
    const unsigned short* __restrict__ A,   // M x K bf16 (row-major)
    const unsigned short* __restrict__ Wt,  // N x K bf16 (pre-transposed)
    const float* __restrict__ bias, const float* __restrict__ resid,
    float* __restrict__ C, unsigned short* __restrict__ Cbf, int N, int K) {
  __shared__ __bf16 As[2][128][40];   // [buf][m][k], 80B row stride (16B align)
  __shared__ __bf16 Bs[2][128][40];   // [buf][n][k]
  const int tid = threadIdx.x;
  const int lane = tid & 31, wid = tid >> 5;
  const int wm = wid & 3, wn = wid >> 2;
  const int row0 = blockIdx.y * 128, col0 = blockIdx.x * 128;

  v8f acc[2][4] = {};

  // Staging: 128x32 bf16 tile = 512 x 16B chunks; each thread copies rows
  // cr and cr+64 for both A and W (4 async b128 per K-step).
  const int cr = tid >> 2, cc = (tid & 3) * 8;
  unsigned as_d[2][2], bs_d[2][2];
#pragma unroll
  for (int p = 0; p < 2; ++p) {
    as_d[p][0] = (unsigned)(size_t)&As[p][cr][cc];
    as_d[p][1] = (unsigned)(size_t)&As[p][64 + cr][cc];
    bs_d[p][0] = (unsigned)(size_t)&Bs[p][cr][cc];
    bs_d[p][1] = (unsigned)(size_t)&Bs[p][64 + cr][cc];
  }
  const unsigned short* aSrc0 = A  + (size_t)(row0 + cr) * K + cc;
  const unsigned short* aSrc1 = A  + (size_t)(row0 + 64 + cr) * K + cc;
  const unsigned short* bSrc0 = Wt + (size_t)(col0 + cr) * K + cc;
  const unsigned short* bSrc1 = Wt + (size_t)(col0 + 64 + cr) * K + cc;

  auto stage = [&](int kb, int p) {
    async_b128(as_d[p][0], aSrc0 + kb);
    async_b128(as_d[p][1], aSrc1 + kb);
    async_b128(bs_d[p][0], bSrc0 + kb);
    async_b128(bs_d[p][1], bSrc1 + kb);
  };

  stage(0, 0);
  async_wait0();
  __syncthreads();

  for (int kb = 0; kb < K; kb += 32) {
    const int cur = (kb >> 5) & 1;
    if (kb + 32 < K) {
      stage(kb + 32, cur ^ 1);  // overlap next-tile copies with WMMA
      __builtin_prefetch(aSrc0 + kb + 64, 0, 1);
    }
    v16bf a0 = load_frag_rm<40>(&As[cur][wm * 32][0], lane);
    v16bf a1 = load_frag_rm<40>(&As[cur][wm * 32 + 16][0], lane);
    v16bf b0 = load_frag_rm<40>(&Bs[cur][wn * 64][0], lane);
    v16bf b1 = load_frag_rm<40>(&Bs[cur][wn * 64 + 16][0], lane);
    v16bf b2 = load_frag_rm<40>(&Bs[cur][wn * 64 + 32][0], lane);
    v16bf b3 = load_frag_rm<40>(&Bs[cur][wn * 64 + 48][0], lane);
    acc[0][0] = wmma_bf16(a0, b0, acc[0][0]);
    acc[0][1] = wmma_bf16(a0, b1, acc[0][1]);
    acc[0][2] = wmma_bf16(a0, b2, acc[0][2]);
    acc[0][3] = wmma_bf16(a0, b3, acc[0][3]);
    acc[1][0] = wmma_bf16(a1, b0, acc[1][0]);
    acc[1][1] = wmma_bf16(a1, b1, acc[1][1]);
    acc[1][2] = wmma_bf16(a1, b2, acc[1][2]);
    acc[1][3] = wmma_bf16(a1, b3, acc[1][3]);
    if (kb + 32 < K) async_wait0();
    __syncthreads();
  }

  const int g = lane >> 4, nl = lane & 15;
#pragma unroll
  for (int mi = 0; mi < 2; ++mi)
#pragma unroll
    for (int ni = 0; ni < 4; ++ni) {
      int c = col0 + wn * 64 + ni * 16 + nl;
      float bv = BIAS ? bias[c] : 0.f;
#pragma unroll
      for (int v = 0; v < 8; ++v) {
        int idx = (row0 + wm * 32 + mi * 16 + g * 8 + v) * N + c;  // int: <2^24
        float val = acc[mi][ni][v];
        if (BIAS) val += bv;
        if (GELU) val = 0.5f * val * (1.f + erff(val * 0.70710678118654752f));
        if (RESID) val += resid[idx];
        if (CF32) C[idx] = val;
        if (CBF) Cbf[idx] = f2bfu(val);
      }
    }
}

// ---------------------------------------------------------------------------
// Flash attention: one block per (b, h, 64-row query tile); 4 waves, each
// owning a 16-row band. Element (t, h, d) of Q lives at
// Q[(b*1024+t)*qld + qoff + h + d*qds]  (handles the (d,k,h) head packing).
// Output: bf16 mirror only, Obf[(b*1024+t)*512 + h*64 + d].
// ---------------------------------------------------------------------------
__global__ __launch_bounds__(128) void attn_kernel(
    const float* __restrict__ Q, int qld, int qoff, int qds,
    const float* __restrict__ KV, int kvld, int koff, int voff, int kvds,
    const float* __restrict__ mask, unsigned short* __restrict__ Obf) {
  __shared__ __bf16 Qs[64][72];     // [t][d], pre-scaled by 1/sqrt(64)
  __shared__ __bf16 Ks[64][72];     // [j][d]
  __shared__ __bf16 Vts[64][72];    // [d][j]  (V transposed)
  __shared__ __bf16 Ps[4][16][72];  // wave-private P bands [row][j]
  __shared__ float mks[64];

  const int tid = threadIdx.x, lane = tid & 31, wid = tid >> 5;
  const int g = lane >> 4, nl = lane & 15;
  const int b = blockIdx.z, h = blockIdx.y, t0 = blockIdx.x * 64;
  const int band0 = wid * 16;

  {  // stage Q tile (pre-scaled), packed b32 LDS stores, int offsets
    int r = tid >> 1, c0 = (tid & 1) * 32;
    const float* qp = Q + (size_t)(b * 1024 + t0 + r) * qld + qoff + h;
#pragma unroll
    for (int d = 0; d < 32; d += 2)
      *(unsigned*)&Qs[r][c0 + d] =
          pack2bf(qp[(c0 + d) * qds] * 0.125f, qp[(c0 + d + 1) * qds] * 0.125f);
  }

  float mq[8];
#pragma unroll
  for (int v = 0; v < 8; ++v)
    mq[v] = mask ? mask[b * 1024 + t0 + band0 + g * 8 + v] : 1.f;

  float mrow[8], lrow[8];
  v8f oacc[4];
  {
    v8f z = {};
#pragma unroll
    for (int nf = 0; nf < 4; ++nf) oacc[nf] = z;
#pragma unroll
    for (int v = 0; v < 8; ++v) { mrow[v] = -3.0e38f; lrow[v] = 0.f; }
  }

  for (int s0 = 0; s0 < 1024; s0 += 64) {
    {  // stage K chunk [j][d] and transposed V chunk [d][j]
      int j = tid >> 1, c0 = (tid & 1) * 32;
      const float* kp = KV + (size_t)(b * 1024 + s0 + j) * kvld + koff + h;
#pragma unroll
      for (int d = 0; d < 32; d += 2)
        *(unsigned*)&Ks[j][c0 + d] =
            pack2bf(kp[(c0 + d) * kvds], kp[(c0 + d + 1) * kvds]);
      int dd = tid >> 1, j0 = (tid & 1) * 32;
      const float* vp = KV + (size_t)(b * 1024 + s0) * kvld + voff + h + dd * kvds;
#pragma unroll
      for (int j2 = 0; j2 < 32; j2 += 2)
        *(unsigned*)&Vts[dd][j0 + j2] =
            pack2bf(vp[(j0 + j2) * kvld], vp[(j0 + j2 + 1) * kvld]);
      if (mask && tid < 64) mks[tid] = mask[b * 1024 + s0 + tid];
    }
    __syncthreads();

    // S = Q * K^T for this wave's 16-row band (kv columns 0..63)
    v16bf qa0 = load_frag_rm<72>(&Qs[band0][0], lane);
    v16bf qa1 = load_frag_rm<72>(&Qs[band0][32], lane);
    v8f sfrag[4];
#pragma unroll
    for (int nf = 0; nf < 4; ++nf) {
      v8f s = {};
      s = wmma_bf16(qa0, load_frag_rm<72>(&Ks[nf * 16][0], lane), s);
      s = wmma_bf16(qa1, load_frag_rm<72>(&Ks[nf * 16][32], lane), s);
      sfrag[nf] = s;
    }

    if (mask) {
#pragma unroll
      for (int nf = 0; nf < 4; ++nf) {
        float mk = mks[nf * 16 + nl];
#pragma unroll
        for (int v = 0; v < 8; ++v)
          if (mq[v] * mk == 0.f) sfrag[nf][v] = -1.0e16f;
      }
    }

    // online softmax per row (rows in VGPR index v per lane-half g; columns
    // spread over 16 lanes of the half + 4 fragments)
#pragma unroll
    for (int v = 0; v < 8; ++v) {
      float mx = sfrag[0][v];
#pragma unroll
      for (int nf = 1; nf < 4; ++nf) mx = fmaxf(mx, sfrag[nf][v]);
#pragma unroll
      for (int off = 1; off < 16; off <<= 1) mx = fmaxf(mx, __shfl_xor(mx, off, 32));
      float mnew = fmaxf(mrow[v], mx);
      float corr = __expf(mrow[v] - mnew);
      float rs = 0.f;
#pragma unroll
      for (int nf = 0; nf < 4; ++nf) {
        float p = __expf(sfrag[nf][v] - mnew);
        sfrag[nf][v] = p;
        rs += p;
        oacc[nf][v] *= corr;
      }
#pragma unroll
      for (int off = 1; off < 16; off <<= 1) rs += __shfl_xor(rs, off, 32);
      lrow[v] = lrow[v] * corr + rs;
      mrow[v] = mnew;
    }

    // C-layout -> A-layout via wave-private LDS bounce, then O += P * V
#pragma unroll
    for (int nf = 0; nf < 4; ++nf)
#pragma unroll
      for (int v = 0; v < 8; ++v)
        Ps[wid][g * 8 + v][nf * 16 + nl] = f2bf(sfrag[nf][v]);

    v16bf pa0 = load_frag_rm<72>(&Ps[wid][0][0], lane);
    v16bf pa1 = load_frag_rm<72>(&Ps[wid][0][32], lane);
#pragma unroll
    for (int nf = 0; nf < 4; ++nf) {
      oacc[nf] = wmma_bf16(pa0, load_frag_rm<72>(&Vts[nf * 16][0], lane), oacc[nf]);
      oacc[nf] = wmma_bf16(pa1, load_frag_rm<72>(&Vts[nf * 16][32], lane), oacc[nf]);
    }
    __syncthreads();
  }

#pragma unroll
  for (int v = 0; v < 8; ++v) {
    float inv = 1.f / lrow[v];
    int r = b * 1024 + t0 + band0 + g * 8 + v;
#pragma unroll
    for (int nf = 0; nf < 4; ++nf)
      Obf[r * 512 + h * 64 + nf * 16 + nl] = f2bfu(oacc[nf][v] * inv);
  }
}

// LayerNorm over D=512, one wave per row; optional bf16 mirror output.
__global__ __launch_bounds__(256) void ln_kernel(
    const float* __restrict__ in, const float* __restrict__ gam,
    const float* __restrict__ bet, float* __restrict__ out,
    unsigned short* __restrict__ outbf, int rows) {
  const int lane = threadIdx.x & 31, wid = threadIdx.x >> 5;
  const int row = blockIdx.x * 8 + wid;
  if (row >= rows) return;
  const float* p = in + (size_t)row * 512;
  float s = 0.f, s2 = 0.f;
  for (int i = lane; i < 512; i += 32) { float v = p[i]; s += v; s2 += v * v; }
#pragma unroll
  for (int off = 16; off >= 1; off >>= 1) {
    s += __shfl_xor(s, off, 32); s2 += __shfl_xor(s2, off, 32);
  }
  float m = s * (1.f / 512.f);
  float var = s2 * (1.f / 512.f) - m * m;
  float inv = rsqrtf(var + 1e-5f);
  for (int i = lane; i < 512; i += 32) {
    float val = (p[i] - m) * inv * gam[i] + bet[i];
    out[row * 512 + i] = val;
    if (outbf) outbf[row * 512 + i] = f2bfu(val);
  }
}

// fp32 -> bf16 bulk convert (4 elems/thread, 8B stores)
__global__ __launch_bounds__(256) void cvt_bf16_kernel(
    const float* __restrict__ in, unsigned short* __restrict__ out) {
  int i = (blockIdx.x * 256 + threadIdx.x) * 4;
  float4 v = *(const float4*)(in + i);
  uint2 o; o.x = pack2bf(v.x, v.y); o.y = pack2bf(v.z, v.w);
  *(uint2*)(out + i) = o;
}

// W[K][N] fp32 -> Wt[N][K] bf16, LDS-tiled 32x32 transpose
__global__ __launch_bounds__(256) void wt_kernel(const float* __restrict__ W,
                                                 unsigned short* __restrict__ Wt,
                                                 int K, int N) {
  __shared__ float tile[32][33];
  const int kb = blockIdx.y * 32, nb = blockIdx.x * 32;
  const int tx = threadIdx.x & 31, ty = threadIdx.x >> 5;
#pragma unroll
  for (int i = 0; i < 32; i += 8)
    tile[ty + i][tx] = W[(kb + ty + i) * N + nb + tx];
  __syncthreads();
#pragma unroll
  for (int i = 0; i < 32; i += 8)
    Wt[(nb + ty + i) * K + kb + tx] = f2bfu(tile[tx][ty + i]);
}

// Wt[n][k] = bf16(Wvk[k][n] + Wvk[k+512][n])  (collapses concat([a,a]) @ Wvk)
__global__ __launch_bounds__(256) void addhalfT_kernel(
    const float* __restrict__ W, unsigned short* __restrict__ Wt) {
  __shared__ float tile[32][33];
  const int kb = blockIdx.y * 32, nb = blockIdx.x * 32;  // k<512, n<1024
  const int tx = threadIdx.x & 31, ty = threadIdx.x >> 5;
#pragma unroll
  for (int i = 0; i < 32; i += 8)
    tile[ty + i][tx] = W[(kb + ty + i) * 1024 + nb + tx] +
                       W[(kb + ty + i + 512) * 1024 + nb + tx];
  __syncthreads();
#pragma unroll
  for (int i = 0; i < 32; i += 8)
    Wt[(nb + ty + i) * 512 + kb + tx] = f2bfu(tile[tx][ty + i]);
}

// ---------------------------------------------------------------------------
extern "C" void kernel_launch(void* const* d_in, const int* in_sizes, int n_in,
                              void* d_out, int out_size, void* d_ws, size_t ws_size,
                              hipStream_t stream) {
  (void)in_sizes; (void)n_in; (void)out_size; (void)ws_size;
  auto P = [&](int i) { return (const float*)d_in[i]; };
  // d_in: 0=x 1=seq 2=mask; x-tag params 3..18; s-tag params 19..34
  const float* x_in = P(0);
  const float* s_in = P(1);
  const float* mask = P(2);

  const size_t E = (size_t)4096 * 512;  // 2,097,152
  float* wsf  = (float*)d_ws;
  float* tBig = wsf;             // 4096 x 2048 fp32 (qkv / kv, read by attn)
  float* tRes = tBig + 4 * E;    // pre-LN residual sum
  float* xCur = tRes + E;
  float* sCur = xCur + E;
  float* tQ   = sCur + E;
  unsigned short* xin_bf   = (unsigned short*)(tQ + E);
  unsigned short* sin_bf   = xin_bf + E;
  unsigned short* tAttn_bf = sin_bf + E;
  unsigned short* xCur_bf  = tAttn_bf + E;
  unsigned short* sCur_bf  = xCur_bf + E;
  unsigned short* tBig_bf  = sCur_bf + E;      // 4096 x 2048 (FFN hidden)
  unsigned short* wtBuf    = tBig_bf + 4 * E;  // up to 2048*512 bf16

  auto wt = [&](const float* W, int K, int N) {
    wt_kernel<<<dim3(N / 32, K / 32), 256, 0, stream>>>(W, wtBuf, K, N);
  };
  auto attn = [&](const float* Qp, int qld, int qoff, int qds, const float* KVp,
                  int kvld, int koff, int voff, int kvds, const float* mk,
                  unsigned short* Obf) {
    attn_kernel<<<dim3(16, 8, 4), 128, 0, stream>>>(Qp, qld, qoff, qds, KVp, kvld,
                                                    koff, voff, kvds, mk, Obf);
  };
  auto ln = [&](const float* in, const float* gg, const float* bb, float* out,
                unsigned short* outbf) {
    ln_kernel<<<dim3(512), 256, 0, stream>>>(in, gg, bb, out, outbf, 4096);
  };
#define GEMM(B_, R_, G_, F_, C_, Abf, bias, resid, Cp, Cbf, N_, K_)                \
  gemm_bf16_kernel<B_, R_, G_, F_, C_><<<dim3((N_) / 128, 32), 256, 0, stream>>>(  \
      Abf, wtBuf, bias, resid, Cp, Cbf, N_, K_)

  cvt_bf16_kernel<<<2048, 256, 0, stream>>>(x_in, xin_bf);
  cvt_bf16_kernel<<<2048, 256, 0, stream>>>(s_in, sin_bf);

  // ---- stage A: self-attention (qkv packing: q@d*24+h, k@+8, v@+16) ----
  wt(P(3), 512, 1536);
  GEMM(false, false, false, true, false, xin_bf, nullptr, nullptr, tBig, nullptr, 1536, 512);
  attn(tBig, 1536, 0, 24, tBig, 1536, 8, 16, 24, nullptr, tAttn_bf);
  wt(P(4), 512, 512);
  GEMM(false, true, false, true, false, tAttn_bf, nullptr, x_in, tRes, nullptr, 512, 512);
  ln(tRes, P(13), P(14), xCur, xCur_bf);              // g1x / be1x

  wt(P(19), 512, 1536);
  GEMM(false, false, false, true, false, sin_bf, nullptr, nullptr, tBig, nullptr, 1536, 512);
  attn(tBig, 1536, 0, 24, tBig, 1536, 8, 16, 24, mask, tAttn_bf);
  wt(P(20), 512, 512);
  GEMM(false, true, false, true, false, tAttn_bf, nullptr, s_in, tRes, nullptr, 512, 512);
  ln(tRes, P(29), P(30), sCur, sCur_bf);              // g1s / be1s

  // ---- stage B: cross-attention (seq-branch uses the updated x) ----
  addhalfT_kernel<<<dim3(32, 16), 256, 0, stream>>>(P(7), wtBuf);  // Wvk_cx
  GEMM(false, false, false, true, false, sCur_bf, nullptr, nullptr, tBig, nullptr, 1024, 512);
  wt(P(5), 512, 512);
  GEMM(true, false, false, true, false, xCur_bf, P(6), nullptr, tQ, nullptr, 512, 512);
  attn(tQ, 512, 0, 8, tBig, 1024, 0, 8, 16, nullptr, tAttn_bf);
  wt(P(8), 512, 512);
  GEMM(false, true, false, true, false, tAttn_bf, nullptr, xCur, tRes, nullptr, 512, 512);
  ln(tRes, P(31), P(32), xCur, xCur_bf);              // g2s / be2s (ref swaps!)

  addhalfT_kernel<<<dim3(32, 16), 256, 0, stream>>>(P(23), wtBuf);  // Wvk_cs
  GEMM(false, false, false, true, false, xCur_bf, nullptr, nullptr, tBig, nullptr, 1024, 512);
  wt(P(21), 512, 512);
  GEMM(true, false, false, true, false, sCur_bf, P(22), nullptr, tQ, nullptr, 512, 512);
  attn(tQ, 512, 0, 8, tBig, 1024, 0, 8, 16, nullptr, tAttn_bf);
  wt(P(24), 512, 512);
  GEMM(false, true, false, true, false, tAttn_bf, nullptr, sCur, tRes, nullptr, 512, 512);
  ln(tRes, P(15), P(16), sCur, sCur_bf);              // g2x / be2x

  // ---- stage C: FFN (exact GELU fused; hidden kept bf16-only) ----
  float* out_x = (float*)d_out;
  float* out_s = out_x + E;
  wt(P(9), 512, 2048);
  GEMM(true, false, true, false, true, xCur_bf, P(10), nullptr, nullptr, tBig_bf, 2048, 512);
  wt(P(11), 2048, 512);
  GEMM(true, true, false, true, false, tBig_bf, P(12), xCur, tRes, nullptr, 512, 2048);
  ln(tRes, P(17), P(18), out_x, nullptr);             // g3x / be3x

  wt(P(25), 512, 2048);
  GEMM(true, false, true, false, true, sCur_bf, P(26), nullptr, nullptr, tBig_bf, 2048, 512);
  wt(P(27), 2048, 512);
  GEMM(true, true, false, true, false, tBig_bf, P(28), sCur, tRes, nullptr, 512, 2048);
  ln(tRes, P(33), P(34), out_s, nullptr);             // g3s / be3s
#undef GEMM
}
